// FOOGD_Module_17609365914252
// MI455X (gfx1250) — compile-verified, matrix operand order checked by reading, not run on hardware
//
#include <hip/hip_runtime.h>

typedef __attribute__((ext_vector_type(16))) __bf16 v16bf;
typedef __attribute__((ext_vector_type(8)))  __bf16 v8bf;
typedef __attribute__((ext_vector_type(8)))  float  v8f;
typedef __attribute__((ext_vector_type(4)))  float  v4f;
typedef int v4i_vs __attribute__((vector_size(16)));   // matches builtin param pointee

#define BATCH 4096
#define DIM   256
#define HID1  512
#define HID2  256
#define HID3  128
#define NBINS 8192
#define BIN_SCALE 128.0f   /* bins cover dist in [0, 64) */
#define LDK   40           /* padded LDS row stride (bf16): 16B aligned, bank-rotating */

#if defined(__HIP_DEVICE_COMPILE__) && __has_builtin(__builtin_amdgcn_global_load_async_to_lds_b128)
#define HAVE_ASYNC_LDS 1
#else
#define HAVE_ASYNC_LDS 0
#endif

// ---------------- deterministic hash RNG (PCG + Box-Muller) ----------------
__device__ __forceinline__ unsigned pcg(unsigned x) {
  x = x * 747796405u + 2891336453u;
  unsigned w = ((x >> ((x >> 28u) + 4u)) ^ x) * 277803737u;
  return (w >> 22u) ^ w;
}
__device__ __forceinline__ float rng_normal(unsigned seed, unsigned idx) {
  unsigned h1 = pcg(idx * 2654435761u + seed * 0x9E3779B9u + 1u);
  unsigned h2 = pcg(h1 ^ 0x85EBCA6Bu);
  float u1 = ((float)h1 + 1.0f) * 2.3283064e-10f;   // (0,1]
  float u2 = (float)h2 * 2.3283064e-10f;
  return sqrtf(-2.0f * logf(u1)) * cosf(6.28318530718f * u2);
}

// 16B global(bf16) -> LDS copy: async (ASYNCcnt) when available, else via VGPRs
__device__ __forceinline__ void stage16(const __bf16* __restrict__ g, __bf16* l) {
#if HAVE_ASYNC_LDS
  __builtin_amdgcn_global_load_async_to_lds_b128(
      (__attribute__((address_space(1))) v4i_vs*)(g),
      (__attribute__((address_space(3))) v4i_vs*)(l), 0, 0);
#else
  *(v8bf*)l = *(const v8bf*)g;
#endif
}
__device__ __forceinline__ void stage_wait() {
#if HAVE_ASYNC_LDS
#if __has_builtin(__builtin_amdgcn_s_wait_asynccnt)
  __builtin_amdgcn_s_wait_asynccnt(0);
#else
  asm volatile("s_wait_asynccnt 0x0" ::: "memory");
#endif
#endif
}

// Fragment load: ISA 16-bit A/B 16x32 layout. Lane half h needs
// k in [8h,8h+8) and [16+8h,16+8h+8): two contiguous v8bf -> 2x ds_load_b128.
__device__ __forceinline__ v16bf load_frag(const __bf16* __restrict__ rowPtr, int hf) {
  const v8bf* p = (const v8bf*)rowPtr;
  v8bf lo = p[hf];
  v8bf hi = p[2 + hf];
  return __builtin_shufflevector(lo, hi, 0,1,2,3,4,5,6,7,8,9,10,11,12,13,14,15);
}

// -------- WMMA GEMM: act(A[MxK](bf16) @ Wt[N][K](bf16) + b) -> Cb (bf16) [+ Cf fp32]
__global__ __launch_bounds__(256) void gemm_bias_act_kernel(
    const __bf16* __restrict__ A, const __bf16* __restrict__ Wt,
    const float* __restrict__ bias, __bf16* __restrict__ Cb,
    float* __restrict__ Cf, int M, int N, int K, int act) {
  __shared__ __attribute__((aligned(16))) __bf16 As[64][LDK];
  __shared__ __attribute__((aligned(16))) __bf16 Bs[128][LDK];
  const int tid = threadIdx.x;
  const int lane = tid & 31, wv = tid >> 5;
  const int wr = wv & 1, wc = wv >> 1;        // 2x4 wave grid, 32x32 per wave
  const int hf = lane >> 4, rr = lane & 15;
  const int rowBase = blockIdx.y * 64;
  const int colBase = blockIdx.x * 128;

  const v8f z8 = {0.f,0.f,0.f,0.f,0.f,0.f,0.f,0.f};
  v8f acc[2][2];
  acc[0][0] = z8; acc[0][1] = z8; acc[1][0] = z8; acc[1][1] = z8;

  const int ar = tid >> 2, ac = tid & 3;      // A chunk: 64 rows x 4 chunks
  for (int k0 = 0; k0 < K; k0 += 32) {
    stage16(&A[(size_t)(rowBase + ar) * K + k0 + ac * 8], &As[ar][ac * 8]);
#pragma unroll
    for (int t = 0; t < 2; ++t) {
      int idx = tid + t * 256, n = idx >> 2, c = idx & 3;
      stage16(&Wt[(size_t)(colBase + n) * K + k0 + c * 8], &Bs[n][c * 8]);
    }
    if (k0 + 32 < K) {
      __builtin_prefetch(&A[(size_t)(rowBase + ar) * K + k0 + 32 + ac * 8], 0, 1);
      __builtin_prefetch(&Wt[(size_t)(colBase + (tid >> 1)) * K + k0 + 32], 0, 1);
    }
    stage_wait();
    __syncthreads();
    v16bf af[2], bfv[2];
    af[0]  = load_frag(&As[wr * 32 + rr][0], hf);
    af[1]  = load_frag(&As[wr * 32 + 16 + rr][0], hf);
    bfv[0] = load_frag(&Bs[wc * 32 + rr][0], hf);
    bfv[1] = load_frag(&Bs[wc * 32 + 16 + rr][0], hf);
#pragma unroll
    for (int i = 0; i < 2; ++i)
#pragma unroll
      for (int j = 0; j < 2; ++j)
        acc[i][j] = __builtin_amdgcn_wmma_f32_16x16x32_bf16(
            false, af[i], false, bfv[j], (short)0, acc[i][j], false, false);
    __syncthreads();
  }
#pragma unroll
  for (int i = 0; i < 2; ++i)
#pragma unroll
    for (int j = 0; j < 2; ++j)
#pragma unroll
      for (int r = 0; r < 8; ++r) {
        int m = rowBase + wr * 32 + i * 16 + r + hf * 8;
        int n = colBase + wc * 32 + j * 16 + rr;
        float v = acc[i][j][r] + bias[n];
        if (act) v = fmaxf(v, 0.0f);
        Cb[(size_t)m * N + n] = (__bf16)v;
        if (Cf) Cf[(size_t)m * N + n] = v;
      }
}

// ------- fused tile kernel over X@Y^T (bf16): mode 0 = distance histogram,
// -------                                      mode 1 = sum of exp(-gamma*sq)
__global__ __launch_bounds__(256) void dist_kernel(
    const __bf16* __restrict__ X, const __bf16* __restrict__ Y,
    const float* __restrict__ nx, const float* __restrict__ ny,
    unsigned* __restrict__ hist, const float* __restrict__ sig3,
    float* __restrict__ sumPtr, int mode) {
  __shared__ __attribute__((aligned(16))) __bf16 As[64][LDK];
  __shared__ __attribute__((aligned(16))) __bf16 Bs[128][LDK];
  __shared__ unsigned lh[NBINS];
  __shared__ float red[256];
  const int tid = threadIdx.x;
  const int lane = tid & 31, wv = tid >> 5;
  const int wr = wv & 1, wc = wv >> 1;
  const int hf = lane >> 4, rr = lane & 15;
  const int rowBase = blockIdx.y * 64;
  const int colBase = blockIdx.x * 128;

  if (mode == 0)
    for (int i = tid; i < NBINS; i += 256) lh[i] = 0u;

  const v8f z8 = {0.f,0.f,0.f,0.f,0.f,0.f,0.f,0.f};
  v8f acc[2][2];
  acc[0][0] = z8; acc[0][1] = z8; acc[1][0] = z8; acc[1][1] = z8;

  const int ar = tid >> 2, ac = tid & 3;
  for (int k0 = 0; k0 < DIM; k0 += 32) {
    stage16(&X[(size_t)(rowBase + ar) * DIM + k0 + ac * 8], &As[ar][ac * 8]);
#pragma unroll
    for (int t = 0; t < 2; ++t) {
      int idx = tid + t * 256, n = idx >> 2, c = idx & 3;
      stage16(&Y[(size_t)(colBase + n) * DIM + k0 + c * 8], &Bs[n][c * 8]);
    }
    if (k0 + 32 < DIM) {
      __builtin_prefetch(&X[(size_t)(rowBase + ar) * DIM + k0 + 32 + ac * 8], 0, 1);
      __builtin_prefetch(&Y[(size_t)(colBase + (tid >> 1)) * DIM + k0 + 32], 0, 1);
    }
    stage_wait();
    __syncthreads();
    v16bf af[2], bfv[2];
    af[0]  = load_frag(&As[wr * 32 + rr][0], hf);
    af[1]  = load_frag(&As[wr * 32 + 16 + rr][0], hf);
    bfv[0] = load_frag(&Bs[wc * 32 + rr][0], hf);
    bfv[1] = load_frag(&Bs[wc * 32 + 16 + rr][0], hf);
#pragma unroll
    for (int i = 0; i < 2; ++i)
#pragma unroll
      for (int j = 0; j < 2; ++j)
        acc[i][j] = __builtin_amdgcn_wmma_f32_16x16x32_bf16(
            false, af[i], false, bfv[j], (short)0, acc[i][j], false, false);
    __syncthreads();
  }

  float gamma = (mode == 1) ? sig3[1] : 0.0f;
  float local = 0.0f;
#pragma unroll
  for (int i = 0; i < 2; ++i)
#pragma unroll
    for (int j = 0; j < 2; ++j)
#pragma unroll
      for (int r = 0; r < 8; ++r) {
        int m = rowBase + wr * 32 + i * 16 + r + hf * 8;
        int n = colBase + wc * 32 + j * 16 + rr;
        float sq = fmaxf(nx[m] + ny[n] - 2.0f * acc[i][j][r], 0.0f);
        if (mode == 0) {
          int b = (int)(sqrtf(sq) * BIN_SCALE);
          if (b > NBINS - 1) b = NBINS - 1;
          atomicAdd(&lh[b], 1u);
        } else {
          local += __expf(-gamma * sq);
        }
      }
  __syncthreads();
  if (mode == 0) {
    for (int i = tid; i < NBINS; i += 256)
      if (lh[i]) atomicAdd(&hist[i], lh[i]);
  } else {
    red[tid] = local;
    __syncthreads();
    for (int s = 128; s > 0; s >>= 1) {
      if (tid < s) red[tid] += red[tid + s];
      __syncthreads();
    }
    if (tid == 0) atomicAdd(sumPtr, red[0]);
  }
}

// ---------------- median from global histogram -> sigma, gamma, 1/sigma^2 ---
__global__ void median_kernel(const unsigned* __restrict__ hist,
                              float* __restrict__ sig3, float total) {
  if (threadIdx.x == 0 && blockIdx.x == 0) {
    float rank = total * 0.5f;
    float cum = 0.0f, sigma = 64.0f;
    for (int b = 0; b < NBINS; ++b) {
      float c = (float)hist[b];
      if (c > 0.0f && cum + c >= rank) {
        sigma = ((float)b + (rank - cum) / c) * (1.0f / BIN_SCALE);
        break;
      }
      cum += c;
    }
    sig3[0] = sigma;
    sig3[1] = 1.0f / (2.0f * sigma * sigma + 1e-8f);
    sig3[2] = 1.0f / (sigma * sigma + 1e-8f);
  }
}

// ------- fused KSD: per 32x64 tile compute FF, SS, SF, FS Grams with WMMA ---
__global__ __launch_bounds__(256) void ksd_kernel(
    const __bf16* __restrict__ F, const __bf16* __restrict__ S,
    const float* __restrict__ nA, const float* __restrict__ sf,
    const float* __restrict__ sig3, float* __restrict__ sumPtr) {
  __shared__ __attribute__((aligned(16))) __bf16 Fa[32][LDK];
  __shared__ __attribute__((aligned(16))) __bf16 Sa[32][LDK];
  __shared__ __attribute__((aligned(16))) __bf16 Fb[64][LDK];
  __shared__ __attribute__((aligned(16))) __bf16 Sb[64][LDK];
  __shared__ float red[256];
  const int tid = threadIdx.x;
  const int lane = tid & 31, wv = tid >> 5;
  const int wr = wv & 1, wc = wv >> 1;   // 2x4 waves, 16x16 tile each
  const int hf = lane >> 4, rr = lane & 15;
  const int rowBase = blockIdx.y * 32;
  const int colBase = blockIdx.x * 64;

  const v8f z8 = {0.f,0.f,0.f,0.f,0.f,0.f,0.f,0.f};
  v8f aFF = z8, aSS = z8, aSF = z8, aFS = z8;

  for (int k0 = 0; k0 < DIM; k0 += 32) {
    {
      int q = tid & 127, r = q >> 2, c = q & 3;
      size_t g = (size_t)(rowBase + r) * DIM + k0 + c * 8;
      if (tid < 128) stage16(&F[g], &Fa[r][c * 8]);
      else           stage16(&S[g], &Sa[r][c * 8]);
    }
#pragma unroll
    for (int t = 0; t < 2; ++t) {
      int idx = tid + t * 256, q = idx & 255, n = q >> 2, c = q & 3;
      size_t g = (size_t)(colBase + n) * DIM + k0 + c * 8;
      if (idx < 256) stage16(&F[g], &Fb[n][c * 8]);
      else           stage16(&S[g], &Sb[n][c * 8]);
    }
    stage_wait();
    __syncthreads();
    v16bf aF = load_frag(&Fa[wr * 16 + rr][0], hf);
    v16bf aS = load_frag(&Sa[wr * 16 + rr][0], hf);
    v16bf bF = load_frag(&Fb[wc * 16 + rr][0], hf);
    v16bf bS = load_frag(&Sb[wc * 16 + rr][0], hf);
    aFF = __builtin_amdgcn_wmma_f32_16x16x32_bf16(false, aF, false, bF, (short)0, aFF, false, false);
    aSS = __builtin_amdgcn_wmma_f32_16x16x32_bf16(false, aS, false, bS, (short)0, aSS, false, false);
    aSF = __builtin_amdgcn_wmma_f32_16x16x32_bf16(false, aS, false, bF, (short)0, aSF, false, false);
    aFS = __builtin_amdgcn_wmma_f32_16x16x32_bf16(false, aF, false, bS, (short)0, aFS, false, false);
    __syncthreads();
  }

  const float gamma = sig3[1], inv_s2 = sig3[2];
  float local = 0.0f;
#pragma unroll
  for (int r = 0; r < 8; ++r) {
    int i = rowBase + wr * 16 + r + hf * 8;
    int j = colBase + wc * 16 + rr;
    float sq = fmaxf(nA[i] + nA[j] - 2.0f * aFF[r], 0.0f);
    float Kv = __expf(-gamma * sq);
    float t1 = aSS[r] * Kv;
    float t2 = (sf[i] - aSF[r]) * Kv * inv_s2;
    float t3 = -(aFS[r] - sf[j]) * Kv * inv_s2;
    float t4 = ((float)DIM * inv_s2 - sq * inv_s2 * inv_s2) * Kv;
    local += t1 + t2 + t3 + t4;
  }
  red[tid] = local;
  __syncthreads();
  for (int s = 128; s > 0; s >>= 1) {
    if (tid < s) red[tid] += red[tid + s];
    __syncthreads();
  }
  if (tid == 0) atomicAdd(sumPtr, red[0]);
}

// ---------------- small helper kernels ----------------
// mode 0: out[row]=||x||^2 ; mode 1: out[row]=||x|| ; mode 2: out[row]=<x,y>
__global__ void row_reduce_kernel(const float* __restrict__ X,
                                  const float* __restrict__ Y,
                                  float* __restrict__ out, int mode) {
  __shared__ float red[256];
  int row = blockIdx.x, tid = threadIdx.x;
  float a = X[(size_t)row * DIM + tid];
  float v = (mode == 2) ? a * Y[(size_t)row * DIM + tid] : a * a;
  red[tid] = v;
  __syncthreads();
  for (int s = 128; s > 0; s >>= 1) {
    if (tid < s) red[tid] += red[tid + s];
    __syncthreads();
  }
  if (tid == 0) out[row] = (mode == 1) ? sqrtf(red[0]) : red[0];
}

// transpose + convert weights: W[K][N] fp32 -> Wt[N][K] bf16
__global__ void convert_wt_kernel(const float* __restrict__ W,
                                  __bf16* __restrict__ Wt, int K, int N) {
  int i = blockIdx.x * 256 + threadIdx.x;
  if (i < K * N) {
    int k = i / N, n = i % N;
    Wt[(size_t)n * K + k] = (__bf16)W[i];
  }
}
// fp32 -> bf16, 8 elements per thread (vectorized)
__global__ void convert_bf_kernel(const float* __restrict__ X,
                                  __bf16* __restrict__ Xb, int n8) {
  int i = blockIdx.x * 256 + threadIdx.x;
  if (i < n8) {
    const v4f* gp = (const v4f*)(X + (size_t)i * 8);
    v4f a = gp[0], b = gp[1];
    v8bf r;
    r[0] = (__bf16)a[0]; r[1] = (__bf16)a[1]; r[2] = (__bf16)a[2]; r[3] = (__bf16)a[3];
    r[4] = (__bf16)b[0]; r[5] = (__bf16)b[1]; r[6] = (__bf16)b[2]; r[7] = (__bf16)b[3];
    *(v8bf*)(Xb + (size_t)i * 8) = r;
  }
}

__global__ void randn_scale_kernel(float* out, __bf16* outb, unsigned seed, float scale, int n) {
  int i = blockIdx.x * 256 + threadIdx.x;
  if (i < n) {
    float v = scale * rng_normal(seed, (unsigned)i);
    out[i] = v;
    if (outb) outb[i] = (__bf16)v;
  }
}
__global__ void add_noise_kernel(const float* a, const float* b, __bf16* c, int n) {
  int i = blockIdx.x * 256 + threadIdx.x;
  if (i < n) c[i] = (__bf16)(a[i] + b[i]);
}
__global__ void langevin_update_kernel(float* z, __bf16* zb, const float* score,
                                       unsigned seed, int n) {
  int i = blockIdx.x * 256 + threadIdx.x;
  if (i < n) {
    float v = z[i] + 0.005f * score[i] + 0.1f * rng_normal(seed, (unsigned)i);
    z[i] = v;
    zb[i] = (__bf16)v;
  }
}
__global__ void dsm_loss_kernel(const float* __restrict__ pred,
                                const float* __restrict__ noise,
                                float* __restrict__ sum, int n) {
  __shared__ float red[256];
  int i = blockIdx.x * 256 + threadIdx.x;
  float d = 0.0f;
  if (i < n) {
    float tru = -noise[i] * 100.0f;  // -noise / sigma_dsm^2
    float e = pred[i] - tru;
    d = e * e;
  }
  red[threadIdx.x] = d;
  __syncthreads();
  for (int s = 128; s > 0; s >>= 1) {
    if (threadIdx.x < s) red[threadIdx.x] += red[threadIdx.x + s];
    __syncthreads();
  }
  if (threadIdx.x == 0) atomicAdd(sum, red[0]);
}
__global__ void zero_f_kernel(float* p, int n) {
  int i = blockIdx.x * 256 + threadIdx.x;
  if (i < n) p[i] = 0.0f;
}
__global__ void zero_u_kernel(unsigned* p, int n) {
  int i = blockIdx.x * 256 + threadIdx.x;
  if (i < n) p[i] = 0u;
}
__global__ void finalize_kernel(const float* __restrict__ s, float* __restrict__ out) {
  if (threadIdx.x == 0 && blockIdx.x == 0) {
    const float invBB = 1.0f / ((float)BATCH * (float)BATCH);
    float l_dsm = 0.5f * (s[15] / (float)(BATCH * DIM)) * 0.01f;
    float l_mmd = (s[12] + s[13] - 2.0f * s[14]) * invBB;
    out[0] = (s[16] * invBB) / (float)DIM;          // ksd_loss
    out[1] = 0.5f * l_dsm + 0.5f * l_mmd;           // sm_loss
  }
}

// ---------------- host orchestration ----------------
extern "C" void kernel_launch(void* const* d_in, const int* in_sizes, int n_in,
                              void* d_out, int out_size, void* d_ws, size_t ws_size,
                              hipStream_t stream) {
  (void)in_sizes; (void)n_in; (void)out_size; (void)ws_size;
  const float* features     = (const float*)d_in[0];
  const float* features_aug = (const float*)d_in[1];
  const float* W1 = (const float*)d_in[2]; const float* b1 = (const float*)d_in[3];
  const float* W2 = (const float*)d_in[4]; const float* b2 = (const float*)d_in[5];
  const float* W3 = (const float*)d_in[6]; const float* b3 = (const float*)d_in[7];
  const float* W4 = (const float*)d_in[8]; const float* b4 = (const float*)d_in[9];
  float* out = (float*)d_out;

  float* w = (float*)d_ws;
  size_t off = 0;
  float* outb  = w + off; off += (size_t)BATCH * DIM;   // fp32 final-layer scores
  float* zz    = w + off; off += (size_t)BATCH * DIM;   // fp32 Langevin state
  float* noise = w + off; off += (size_t)BATCH * DIM;
  float* nF    = w + off; off += BATCH;
  float* nA    = w + off; off += BATCH;
  float* nZ    = w + off; off += BATCH;
  float* sfv   = w + off; off += BATCH;
  float* scal  = w + off; off += 32;
  unsigned* hist = (unsigned*)(w + off); off += NBINS;
  __bf16* w1t  = (__bf16*)(w + off); off += (size_t)HID1 * DIM  / 2;
  __bf16* w2t  = (__bf16*)(w + off); off += (size_t)HID2 * HID1 / 2;
  __bf16* w3t  = (__bf16*)(w + off); off += (size_t)HID3 * HID2 / 2;
  __bf16* w4t  = (__bf16*)(w + off); off += (size_t)DIM  * HID3 / 2;
  __bf16* fb   = (__bf16*)(w + off); off += (size_t)BATCH * DIM  / 2;  // features bf16
  __bf16* ab   = (__bf16*)(w + off); off += (size_t)BATCH * DIM  / 2;  // features_aug bf16
  __bf16* xinb = (__bf16*)(w + off); off += (size_t)BATCH * DIM  / 2;
  __bf16* zb   = (__bf16*)(w + off); off += (size_t)BATCH * DIM  / 2;
  __bf16* h1b  = (__bf16*)(w + off); off += (size_t)BATCH * HID1 / 2;
  __bf16* h2b  = (__bf16*)(w + off); off += (size_t)BATCH * HID2 / 2;
  __bf16* h3b  = (__bf16*)(w + off); off += (size_t)BATCH * HID3 / 2;
  __bf16* outbb= (__bf16*)(w + off); off += (size_t)BATCH * DIM  / 2;

  auto gemm = [&](const __bf16* A, const __bf16* Wt, const float* bi,
                  __bf16* Cb, float* Cf, int M, int N, int K, int act) {
    gemm_bias_act_kernel<<<dim3(N / 128, M / 64), 256, 0, stream>>>(A, Wt, bi, Cb, Cf, M, N, K, act);
  };
  auto mlp = [&](const __bf16* x) {   // fp32 scores -> outb, bf16 -> outbb
    gemm(x,   w1t, b1, h1b, nullptr, BATCH, HID1, DIM,  1);
    gemm(h1b, w2t, b2, h2b, nullptr, BATCH, HID2, HID1, 1);
    gemm(h2b, w3t, b3, h3b, nullptr, BATCH, HID3, HID2, 1);
    gemm(h3b, w4t, b4, outbb, outb,  BATCH, DIM,  HID3, 0);
  };

  const int ND = BATCH * DIM;
  const float totalBB = (float)BATCH * (float)BATCH;
  zero_f_kernel<<<1, 256, 0, stream>>>(scal, 32);

  // ---- one-time conversions to bf16 ----
  convert_wt_kernel<<<(DIM  * HID1 + 255) / 256, 256, 0, stream>>>(W1, w1t, DIM,  HID1);
  convert_wt_kernel<<<(HID1 * HID2 + 255) / 256, 256, 0, stream>>>(W2, w2t, HID1, HID2);
  convert_wt_kernel<<<(HID2 * HID3 + 255) / 256, 256, 0, stream>>>(W3, w3t, HID2, HID3);
  convert_wt_kernel<<<(HID3 * DIM  + 255) / 256, 256, 0, stream>>>(W4, w4t, HID3, DIM);
  convert_bf_kernel<<<(ND / 8 + 255) / 256, 256, 0, stream>>>(features, fb, ND / 8);
  convert_bf_kernel<<<(ND / 8 + 255) / 256, 256, 0, stream>>>(features_aug, ab, ND / 8);

  // ---- OOD scores: ||mlp(features)|| per row ----
  mlp(fb);
  row_reduce_kernel<<<BATCH, 256, 0, stream>>>(outb, outb, out + 2, 1);

  // ---- DSM loss ----
  randn_scale_kernel<<<ND / 256, 256, 0, stream>>>(noise, nullptr, 42u, 0.1f, ND);
  add_noise_kernel<<<ND / 256, 256, 0, stream>>>(features, noise, xinb, ND);
  mlp(xinb);
  dsm_loss_kernel<<<ND / 256, 256, 0, stream>>>(outb, noise, scal + 15, ND);

  // ---- Langevin sampling (10 steps) ----
  randn_scale_kernel<<<ND / 256, 256, 0, stream>>>(zz, zb, 43u, 1.0f, ND);
  for (int t = 0; t < 10; ++t) {
    mlp(zb);
    langevin_update_kernel<<<ND / 256, 256, 0, stream>>>(zz, zb, outb, 1000u + (unsigned)t, ND);
  }

  // ---- row squared norms ----
  row_reduce_kernel<<<BATCH, 256, 0, stream>>>(features, features, nF, 0);
  row_reduce_kernel<<<BATCH, 256, 0, stream>>>(zz, zz, nZ, 0);
  row_reduce_kernel<<<BATCH, 256, 0, stream>>>(features_aug, features_aug, nA, 0);

  dim3 dgrid(BATCH / 128, BATCH / 64);
  // ---- K_xx ----
  zero_u_kernel<<<NBINS / 256, 256, 0, stream>>>(hist, NBINS);
  dist_kernel<<<dgrid, 256, 0, stream>>>(fb, fb, nF, nF, hist, scal + 0, scal + 12, 0);
  median_kernel<<<1, 32, 0, stream>>>(hist, scal + 0, totalBB);
  dist_kernel<<<dgrid, 256, 0, stream>>>(fb, fb, nF, nF, hist, scal + 0, scal + 12, 1);
  // ---- K_yy ----
  zero_u_kernel<<<NBINS / 256, 256, 0, stream>>>(hist, NBINS);
  dist_kernel<<<dgrid, 256, 0, stream>>>(zb, zb, nZ, nZ, hist, scal + 3, scal + 13, 0);
  median_kernel<<<1, 32, 0, stream>>>(hist, scal + 3, totalBB);
  dist_kernel<<<dgrid, 256, 0, stream>>>(zb, zb, nZ, nZ, hist, scal + 3, scal + 13, 1);
  // ---- K_xy ----
  zero_u_kernel<<<NBINS / 256, 256, 0, stream>>>(hist, NBINS);
  dist_kernel<<<dgrid, 256, 0, stream>>>(fb, zb, nF, nZ, hist, scal + 6, scal + 14, 0);
  median_kernel<<<1, 32, 0, stream>>>(hist, scal + 6, totalBB);
  dist_kernel<<<dgrid, 256, 0, stream>>>(fb, zb, nF, nZ, hist, scal + 6, scal + 14, 1);

  // ---- KSD on augmented features ----
  mlp(ab);  // scores: fp32 in outb, bf16 in outbb
  row_reduce_kernel<<<BATCH, 256, 0, stream>>>(outb, features_aug, sfv, 2);
  zero_u_kernel<<<NBINS / 256, 256, 0, stream>>>(hist, NBINS);
  dist_kernel<<<dgrid, 256, 0, stream>>>(ab, ab, nA, nA, hist, scal + 9, scal + 16, 0);
  median_kernel<<<1, 32, 0, stream>>>(hist, scal + 9, totalBB);
  ksd_kernel<<<dim3(BATCH / 64, BATCH / 32), 256, 0, stream>>>(ab, outbb, nA, sfv, scal + 9, scal + 16);

  finalize_kernel<<<1, 32, 0, stream>>>(scal, out);
}
